// Model_5875515261085
// MI455X (gfx1250) — matrix operand order repaired and checked by reading, not compiled, not run on hardware
//
#include <hip/hip_runtime.h>

// CDNA5 / gfx1250 implementation of the ptensors GNN forward pass.
// GEMMs use V_WMMA_F32_16X16X32_BF16 (bf16 in, f32 accumulate). Weights are
// pre-packed into the WMMA B-operand lane layout (2x b128 per fragment);
// activations are BN-normalized into zero-padded bf16 row-major (K padded to
// 64, N padded to 32). Each wave computes a 16x32 output strip: one A
// fragment feeds two WMMAs (2-way N blocking), K-loop unrolled x2.

typedef __bf16 bf16;
typedef __attribute__((ext_vector_type(16))) __bf16 v16bf;
typedef __attribute__((ext_vector_type(8)))  __bf16 v8bf;
typedef __attribute__((ext_vector_type(8)))  float  v8f;

#define TPB 256
static inline int a16(int x) { return (x + 15) & ~15; }
static inline int a32i(int x) { return (x + 31) & ~31; }
static inline int a64(int x) { return (x + 63) & ~63; }

// ---------------- embedding: h = x @ emb_w + emb_b  ([3072,11]x[11,64]) ----
__global__ void k_embed(const float* __restrict__ x, const float* __restrict__ w,
                        const float* __restrict__ b, float* __restrict__ h) {
  size_t idx = (size_t)blockIdx.x * blockDim.x + threadIdx.x;
  if (idx >= (size_t)3072 * 64) return;
  int r = (int)(idx >> 6), c = (int)(idx & 63);
  float acc = b[c];
#pragma unroll
  for (int k = 0; k < 11; ++k) acc += x[r * 11 + k] * w[k * 64 + c];
  h[idx] = acc;
}

// ---------------- scatter-add: out[idx[r], c] += X[r, c] -------------------
__global__ void k_scatter(const float* __restrict__ X, int ld,
                          const int* __restrict__ idx, int Rin, int C,
                          float* __restrict__ out) {
  size_t t = (size_t)blockIdx.x * blockDim.x + threadIdx.x;
  if (t >= (size_t)Rin * C) return;
  int r = (int)(t / C), c = (int)(t % C);
  int n = idx ? idx[r] : r;
  atomicAdd(&out[(size_t)n * C + c], X[(size_t)r * ld + c]);
}

// ---------------- Bv[tdom[r], c] += S[tatom[r], c] -------------------------
__global__ void k_gather_scatter(const float* __restrict__ S,
                                 const int* __restrict__ tdom,
                                 const int* __restrict__ tatom, int Rk, int C,
                                 float* __restrict__ Bv) {
  size_t t = (size_t)blockIdx.x * blockDim.x + threadIdx.x;
  if (t >= (size_t)Rk * C) return;
  int r = (int)(t / C), c = (int)(t % C);
  atomicAdd(&Bv[(size_t)tdom[r] * C + c], S[(size_t)tatom[r] * C + c]);
}

__global__ void k_count(const int* __restrict__ dom, int Rk, float* __restrict__ cnt) {
  int r = blockIdx.x * blockDim.x + threadIdx.x;
  if (r < Rk) atomicAdd(&cnt[dom[r]], 1.0f);
}

__global__ void k_div_cnt(float* __restrict__ S, const float* __restrict__ cnt,
                          int N, int C) {
  size_t t = (size_t)blockIdx.x * blockDim.x + threadIdx.x;
  if (t >= (size_t)N * C) return;
  int n = (int)(t / C);
  float d = cnt[n];
  S[t] = (d > 0.0f) ? S[t] / d : 0.0f;
}

// ---- BN stats over the virtual concat [Bv[tdom], S[tatom]] (2C channels) --
__global__ void k_bn_stats_cat(const float* __restrict__ Bv, const float* __restrict__ S,
                               const int* __restrict__ tdom, const int* __restrict__ tatom,
                               int Rk, int C, const float* __restrict__ g,
                               const float* __restrict__ bp,
                               float* __restrict__ scale, float* __restrict__ shift) {
  int j = blockIdx.x;  // 0 .. 2C-1
  __shared__ float ssum[TPB], ssq[TPB];
  float s = 0.f, q = 0.f;
  for (int r = threadIdx.x; r < Rk; r += blockDim.x) {
    float v = (j < C) ? Bv[(size_t)tdom[r] * C + j] : S[(size_t)tatom[r] * C + (j - C)];
    s += v; q += v * v;
  }
  ssum[threadIdx.x] = s; ssq[threadIdx.x] = q;
  __syncthreads();
  for (int st = TPB / 2; st > 0; st >>= 1) {
    if (threadIdx.x < st) { ssum[threadIdx.x] += ssum[threadIdx.x + st];
                            ssq[threadIdx.x]  += ssq[threadIdx.x + st]; }
    __syncthreads();
  }
  if (threadIdx.x == 0) {
    float m = ssum[0] / (float)Rk;
    float v = ssq[0] / (float)Rk - m * m;           // biased variance (ddof=0)
    float sc = g[j] * rsqrtf(v + 1e-5f);
    scale[j] = sc; shift[j] = bp[j] - m * sc;
  }
}

// ---- BN stats over a plain f32 matrix [R x ld], channel j = blockIdx.x ----
__global__ void k_bn_stats_mat(const float* __restrict__ X, int ld, int R,
                               const float* __restrict__ g, const float* __restrict__ bp,
                               float* __restrict__ scale, float* __restrict__ shift) {
  int j = blockIdx.x;
  __shared__ float ssum[TPB], ssq[TPB];
  float s = 0.f, q = 0.f;
  for (int r = threadIdx.x; r < R; r += blockDim.x) {
    float v = X[(size_t)r * ld + j];
    s += v; q += v * v;
  }
  ssum[threadIdx.x] = s; ssq[threadIdx.x] = q;
  __syncthreads();
  for (int st = TPB / 2; st > 0; st >>= 1) {
    if (threadIdx.x < st) { ssum[threadIdx.x] += ssum[threadIdx.x + st];
                            ssq[threadIdx.x]  += ssq[threadIdx.x + st]; }
    __syncthreads();
  }
  if (threadIdx.x == 0) {
    float m = ssum[0] / (float)R;
    float v = ssq[0] / (float)R - m * m;
    float sc = g[j] * rsqrtf(v + 1e-5f);
    scale[j] = sc; shift[j] = bp[j] - m * sc;
  }
}

// ---- build zero-padded bf16 A-operand from gathered concat + fused BN -----
__global__ void k_norm1_bf16(const float* __restrict__ Bv, const float* __restrict__ S,
                             const int* __restrict__ tdom, const int* __restrict__ tatom,
                             int Rk, int Rpad, int C, int Kpad,
                             const float* __restrict__ scale, const float* __restrict__ shift,
                             bf16* __restrict__ out) {
  size_t t = (size_t)blockIdx.x * blockDim.x + threadIdx.x;
  if (t >= (size_t)Rpad * Kpad) return;
  int r = (int)(t / Kpad), j = (int)(t % Kpad);
  float v = 0.f;
  if (r < Rk && j < 2 * C) {
    float x = (j < C) ? Bv[(size_t)tdom[r] * C + j] : S[(size_t)tatom[r] * C + (j - C)];
    v = x * scale[j] + shift[j];
  }
  out[t] = (bf16)v;
}

// ---- f32 matrix -> zero-padded bf16, optional fused BN --------------------
__global__ void k_cvt_bf16(const float* __restrict__ X, int ld, int Rk, int Rpad,
                           int Cv, int Kpad, const float* __restrict__ scale,
                           const float* __restrict__ shift, int use_bn,
                           bf16* __restrict__ out) {
  size_t t = (size_t)blockIdx.x * blockDim.x + threadIdx.x;
  if (t >= (size_t)Rpad * Kpad) return;
  int r = (int)(t / Kpad), j = (int)(t % Kpad);
  float v = 0.f;
  if (r < Rk && j < Cv) {
    v = X[(size_t)r * ld + j];
    if (use_bn) v = v * scale[j] + shift[j];
  }
  out[t] = (bf16)v;
}

// ---- weight f32 [K x N] -> bf16 packed WMMA B-operand layout --------------
// out[((kt*Nt + nt)*32 + lane)*16 + i] = W[kt*32 + (lane>=16?16:0) + i]
//                                        [nt*16 + (lane&15)]
__global__ void k_wcvt_packB(const float* __restrict__ W, int Kv, int Nv,
                             int Kpad, int Nt, bf16* __restrict__ out) {
  size_t t = (size_t)blockIdx.x * blockDim.x + threadIdx.x;
  if (t >= (size_t)Kpad * Nt * 16) return;
  int i    = (int)(t & 15);
  int lane = (int)((t >> 4) & 31);
  int tile = (int)(t >> 9);
  int nt = tile % Nt;
  int kt = tile / Nt;
  int k = kt * 32 + ((lane >= 16) ? 16 : 0) + i;
  int n = nt * 16 + (lane & 15);
  out[t] = (bf16)((k < Kv && n < Nv) ? W[(size_t)k * Nv + n] : 0.0f);
}

// ---- WMMA GEMM: C = relu?(A[MxK] @ B[KxN] + bias) -------------------------
// A: bf16 row-major, rows padded to 16, K padded to 64 (lda = Kpad, zeros).
// Bp: packed fragments [Kpad/32][Nt][32 lanes][16]. One wave computes a
// 16x32 strip (two N tiles share one A fragment); K-loop unrolled x2 ->
// 4 v_wmma per iteration. Nt is even (N padded to 32).
__global__ void __launch_bounds__(TPB)
k_gemm_bf16(const bf16* __restrict__ A, int lda, const v16bf* __restrict__ Bp,
            int Nt, const float* __restrict__ bias, float* __restrict__ Cm,
            int ldc, int M, int N, int K, int relu) {
  int lane  = threadIdx.x & 31;
  int wave  = threadIdx.x >> 5;
  int mtile = blockIdx.y * 8 + wave;
  int Mpad  = (M + 15) & ~15;
  if (mtile * 16 >= Mpad) return;          // wave-uniform exit; EXEC stays full
  int row0  = mtile * 16;
  int nt0   = blockIdx.x * 2;
  int m16   = lane & 15;
  int hi    = (lane >= 16) ? 1 : 0;
  int col0  = nt0 * 16 + m16;
  int col1  = col0 + 16;

  v8f acc0, acc1;
  float bv0 = (col0 < N) ? bias[col0] : 0.0f;
  float bv1 = (col1 < N) ? bias[col1] : 0.0f;
#pragma unroll
  for (int p = 0; p < 8; ++p) { acc0[p] = bv0; acc1[p] = bv1; }

  int Kpad = (K + 63) & ~63;               // lda == Kpad (multiple of 64)
  const bf16* Arow = A + (size_t)(row0 + m16) * lda;
  for (int k0 = 0; k0 < Kpad; k0 += 64) {
    int kt = k0 >> 5;
#pragma unroll
    for (int u = 0; u < 2; ++u) {
      // A 16x32 frag: lanes<16 hold K 0-15, lanes>=16 hold K 16-31
      v8bf a0 = *(const v8bf*)(Arow + k0 + u * 32 + hi * 16);
      v8bf a1 = *(const v8bf*)(Arow + k0 + u * 32 + hi * 16 + 8);
      v16bf a = __builtin_shufflevector(a0, a1, 0, 1, 2, 3, 4, 5, 6, 7,
                                        8, 9, 10, 11, 12, 13, 14, 15);
      const v16bf* brow = Bp + ((size_t)(kt + u) * Nt + nt0) * 32 + lane;
      v16bf b0 = brow[0];
      v16bf b1 = brow[32];
      acc0 = __builtin_amdgcn_wmma_f32_16x16x32_bf16(false, a, false, b0,
                                                     (short)0, acc0, false, false);
      acc1 = __builtin_amdgcn_wmma_f32_16x16x32_bf16(false, a, false, b1,
                                                     (short)0, acc1, false, false);
    }
  }
#pragma unroll
  for (int p = 0; p < 8; ++p) {            // C 16x16: M = p + 8*hi
    int r = row0 + p + (hi ? 8 : 0);
    if (r < M) {
      if (col0 < N) {
        float v = acc0[p];
        if (relu) v = fmaxf(v, 0.0f);
        Cm[(size_t)r * ldc + col0] = v;
      }
      if (col1 < N) {
        float v = acc1[p];
        if (relu) v = fmaxf(v, 0.0f);
        Cm[(size_t)r * ldc + col1] = v;
      }
    }
  }
}

// params pytree leaves in jax tree order (dict keys sorted alphabetically)
static const int kLeafSizes[40] = {
    600, 600,                                        // bn_b, bn_g
    64, 300, 128, 128, 64, 64, 8192, 19200,          // c1: b1,b2,bn1_b,bn1_g,bn2_b,bn2_g,w1,w2
    300, 300, 600, 600, 300, 300, 180000, 90000,     // c2
    300, 300, 600, 600, 300, 300, 180000, 90000,     // c3
    300, 600, 600, 600, 300, 300, 180000, 180000,    // c4
    64, 704,                                         // emb_b, emb_w
    600, 360000, 600, 360000                         // l1_b, l1_w, l2_b, l2_w
};

#define L1D(n, kern, ...)                                                   \
  do {                                                                      \
    size_t _n = (size_t)(n);                                                \
    unsigned _g = (unsigned)((_n + TPB - 1) / TPB);                         \
    kern<<<dim3(_g), dim3(TPB), 0, stream>>>(__VA_ARGS__);                  \
  } while (0)

extern "C" void kernel_launch(void* const* d_in, const int* in_sizes, int n_in,
                              void* d_out, int out_size, void* d_ws, size_t ws_size,
                              hipStream_t stream) {
  if (n_in < 11) return;
  const float* x = (const float*)d_in[0];

  const float* leaves[40];
  int gbase;
  if (n_in >= 50) {                       // params passed as 40 separate leaves
    for (int i = 0; i < 40; ++i) leaves[i] = (const float*)d_in[1 + i];
    gbase = 41;
  } else {                                // params passed as one flat blob
    const float* blob = (const float*)d_in[1];
    size_t off = 0;
    for (int i = 0; i < 40; ++i) { leaves[i] = blob + off; off += kLeafSizes[i]; }
    gbase = 2;
  }
  const int* dom[4]; const int* atm[4]; int R[4];
  for (int k = 0; k < 4; ++k) {
    dom[k] = (const int*)d_in[gbase + 2 * k];
    atm[k] = (const int*)d_in[gbase + 2 * k + 1];
    R[k]   = in_sizes[gbase + 2 * k];
  }
  const int* batch = (const int*)d_in[gbase + 8];

  const float* bn_b = leaves[0];
  const float* bn_g = leaves[1];
  struct ConvP { const float *b1, *b2, *bn1_b, *bn1_g, *bn2_b, *bn2_g, *w1, *w2; } cp[4];
  for (int L = 0; L < 4; ++L) {
    const float* const* q = &leaves[2 + L * 8];
    cp[L].b1 = q[0]; cp[L].b2 = q[1]; cp[L].bn1_b = q[2]; cp[L].bn1_g = q[3];
    cp[L].bn2_b = q[4]; cp[L].bn2_g = q[5]; cp[L].w1 = q[6]; cp[L].w2 = q[7];
  }
  const float* emb_b = leaves[34];
  const float* emb_w = leaves[35];
  const float* l1_b = leaves[36]; const float* l1_w = leaves[37];
  const float* l2_b = leaves[38]; const float* l2_w = leaves[39];

  // layer dims: K pads to 64 (x2-unrolled WMMA loop), N pads to 32 (2-way
  // N-tile blocking per wave -> even 16-col tile count)
  const int Cin[4]  = {64, 300, 300, 300};
  const int Cmid[4] = {64, 300, 300, 300};
  const int Co[4]   = {300, 300, 300, 600};
  int K1p[4], N1p[4], K2p[4], N2p[4], Rp[4];
  for (int L = 0; L < 4; ++L) {
    K1p[L] = a64(2 * Cin[L]); N1p[L] = a32i(Cmid[L]);
    K2p[L] = a64(Cmid[L]);    N2p[L] = a32i(Co[L]);
    Rp[L]  = a16(R[L]);
  }
  const int KHp = a64(600);    // 640: head GEMM K pad
  const int NHp = a32i(600);   // 608: head GEMM N pad

  // ---- workspace bump allocator ----
  size_t off = 0;
  char* base = (char*)d_ws;
  auto alloc = [&](size_t bytes) -> void* {
    void* p = base + off;
    off = (off + bytes + 255) & ~(size_t)255;
    return p;
  };
  float* scale = (float*)alloc(1280 * 4);
  float* shift = (float*)alloc(1280 * 4);
  float* H0    = (float*)alloc((size_t)3072 * 64 * 4);
  float* S     = (float*)alloc((size_t)3072 * 600 * 4);
  float* Bv    = (float*)alloc((size_t)3072 * 600 * 4);
  float* cnt   = (float*)alloc(3072 * 4);
  float* gpool = (float*)alloc((size_t)40 * 600 * 4);
  float* F1    = (float*)alloc((size_t)48 * NHp * 4);
  float* F2    = (float*)alloc((size_t)48 * NHp * 4);
  bf16*  gbf   = (bf16*)alloc((size_t)48 * KHp * 2);
  bf16*  w1bf[4]; bf16* w2bf[4];
  for (int L = 0; L < 4; ++L) {
    w1bf[L] = (bf16*)alloc((size_t)K1p[L] * N1p[L] * 2);
    w2bf[L] = (bf16*)alloc((size_t)K2p[L] * N2p[L] * 2);
  }
  bf16* l1bf = (bf16*)alloc((size_t)KHp * NHp * 2);
  bf16* l2bf = (bf16*)alloc((size_t)KHp * NHp * 2);
  size_t abfMax = 0, g1Max = 0;
  for (int L = 0; L < 4; ++L) {
    size_t s1 = (size_t)Rp[L] * K1p[L] * 2; if (s1 > abfMax) abfMax = s1;
    size_t s2 = (size_t)Rp[L] * K2p[L] * 2; if (s2 > abfMax) abfMax = s2;
    size_t s3 = (size_t)Rp[L] * N1p[L] * 4; if (s3 > g1Max) g1Max = s3;
  }
  bf16*  Abf = (bf16*)alloc(abfMax);          // reused for both GEMM inputs per layer
  float* G1  = (float*)alloc(g1Max);
  size_t hb0sz = (size_t)Rp[0] * N2p[0] * 4;
  { size_t s = (size_t)Rp[2] * N2p[2] * 4; if (s > hb0sz) hb0sz = s; }
  size_t hb1sz = (size_t)Rp[1] * N2p[1] * 4;
  { size_t s = (size_t)Rp[3] * N2p[3] * 4; if (s > hb1sz) hb1sz = s; }
  float* Hb[2];
  Hb[0] = (float*)alloc(hb0sz);
  Hb[1] = (float*)alloc(hb1sz);
  if (off > ws_size) return;                  // workspace too small: bail deterministically

  auto gemm = [&](const bf16* A, int lda, const bf16* Bp, int Npad,
                  const float* bias, float* Cm, int ldc, int M, int Nv, int K,
                  int relu) {
    int Mpad = a16(M);
    dim3 grid((unsigned)(Npad / 32), (unsigned)((Mpad / 16 + 7) / 8));
    k_gemm_bf16<<<grid, dim3(TPB), 0, stream>>>(A, lda, (const v16bf*)Bp,
                                                Npad / 16, bias, Cm, ldc, M, Nv,
                                                K, relu);
  };

  // ---- convert weights to packed bf16 B-fragments ----
  for (int L = 0; L < 4; ++L) {
    L1D((size_t)K1p[L] * N1p[L], k_wcvt_packB, cp[L].w1, 2 * Cin[L], Cmid[L],
        K1p[L], N1p[L] / 16, w1bf[L]);
    L1D((size_t)K2p[L] * N2p[L], k_wcvt_packB, cp[L].w2, Cmid[L], Co[L],
        K2p[L], N2p[L] / 16, w2bf[L]);
  }
  L1D((size_t)KHp * NHp, k_wcvt_packB, l1_w, 600, 600, KHp, NHp / 16, l1bf);
  L1D((size_t)KHp * NHp, k_wcvt_packB, l2_w, 600, 600, KHp, NHp / 16, l2bf);

  // ---- embedding ----
  L1D((size_t)3072 * 64, k_embed, x, emb_w, emb_b, H0);

  // ---- conv layers ----
  const float* Hin = H0; int ldH = 64, Rin = 3072;
  const int* srcA = nullptr;   // layer 1 src atoms are the identity
  for (int L = 0; L < 4; ++L) {
    int C = Cin[L];
    hipMemsetAsync(S, 0, (size_t)3072 * C * 4, stream);
    hipMemsetAsync(Bv, 0, (size_t)3072 * C * 4, stream);
    L1D((size_t)Rin * C, k_scatter, Hin, ldH, srcA, Rin, C, S);
    L1D((size_t)R[L] * C, k_gather_scatter, S, dom[L], atm[L], R[L], C, Bv);
    k_bn_stats_cat<<<dim3(2 * C), dim3(TPB), 0, stream>>>(
        Bv, S, dom[L], atm[L], R[L], C, cp[L].bn1_g, cp[L].bn1_b, scale, shift);
    L1D((size_t)Rp[L] * K1p[L], k_norm1_bf16, Bv, S, dom[L], atm[L],
        R[L], Rp[L], C, K1p[L], scale, shift, Abf);
    gemm(Abf, K1p[L], w1bf[L], N1p[L], cp[L].b1, G1, N1p[L], R[L], Cmid[L],
         2 * C, 1);
    k_bn_stats_mat<<<dim3(Cmid[L]), dim3(TPB), 0, stream>>>(
        G1, N1p[L], R[L], cp[L].bn2_g, cp[L].bn2_b, scale, shift);
    L1D((size_t)Rp[L] * K2p[L], k_cvt_bf16, G1, N1p[L], R[L], Rp[L], Cmid[L],
        K2p[L], scale, shift, 1, Abf);
    float* Hout = Hb[L & 1];
    gemm(Abf, K2p[L], w2bf[L], N2p[L], cp[L].b2, Hout, N2p[L], R[L], Co[L],
         Cmid[L], 1);
    Hin = Hout; ldH = N2p[L]; Rin = R[L]; srcA = atm[L];
  }

  // ---- readout: mean over atoms per domain, sum per graph ----
  hipMemsetAsync(S, 0, (size_t)3072 * 600 * 4, stream);
  hipMemsetAsync(cnt, 0, 3072 * 4, stream);
  hipMemsetAsync(gpool, 0, (size_t)40 * 600 * 4, stream);
  L1D((size_t)R[3] * 600, k_scatter, Hin, ldH, dom[3], R[3], 600, S);
  L1D(R[3], k_count, dom[3], R[3], cnt);
  L1D((size_t)3072 * 600, k_div_cnt, S, cnt, 3072, 600);
  L1D((size_t)3072 * 600, k_scatter, S, 600, batch, 3072, 600, gpool);

  // ---- head: BN -> relu(l1) -> relu(l2) -> l2 ----
  k_bn_stats_mat<<<dim3(600), dim3(TPB), 0, stream>>>(gpool, 600, 40, bn_g,
                                                      bn_b, scale, shift);
  L1D((size_t)48 * KHp, k_cvt_bf16, gpool, 600, 40, 48, 600, KHp, scale, shift,
      1, gbf);
  gemm(gbf, KHp, l1bf, NHp, l1_b, F1, NHp, 40, 600, 600, 1);
  L1D((size_t)48 * KHp, k_cvt_bf16, F1, NHp, 40, 48, 600, KHp,
      (const float*)nullptr, (const float*)nullptr, 0, gbf);
  gemm(gbf, KHp, l2bf, NHp, l2_b, F2, NHp, 40, 600, 600, 1);
  L1D((size_t)48 * KHp, k_cvt_bf16, F2, NHp, 40, 48, 600, KHp,
      (const float*)nullptr, (const float*)nullptr, 0, gbf);
  gemm(gbf, KHp, l2bf, NHp, l2_b, (float*)d_out, 600, 40, 600, 600, 0);
  (void)out_size;
}
